// VCQ_66795331387506
// MI455X (gfx1250) — matrix operand order repaired and checked by reading, not compiled, not run on hardware
//
#include <hip/hip_runtime.h>
#include <hip/hip_bf16.h>

// ---------------------------------------------------------------------------
// MI455X (gfx1250) VCQ forward. All contractions via v_wmma_f32_16x16x32_bf16.
// Linear A-tiles and NN B-tiles are staged into LDS by the Tensor Data Mover
// (tensor_load_to_lds + s_wait_tensorcnt), double-buffered, 1 barrier/K-step.
// ---------------------------------------------------------------------------

typedef __attribute__((ext_vector_type(16))) __bf16 v16bf;
typedef __attribute__((ext_vector_type(8)))  __bf16 v8bf;
typedef __attribute__((ext_vector_type(4)))  __bf16 v4bf;
typedef __attribute__((ext_vector_type(8)))  float  v8f;
typedef __attribute__((ext_vector_type(4)))  unsigned u32x4;
typedef __attribute__((ext_vector_type(8)))  int     i32x8;
typedef __attribute__((ext_vector_type(4)))  int     i32x4;

#if defined(__has_builtin)
#if __has_builtin(__builtin_amdgcn_tensor_load_to_lds) && __has_builtin(__builtin_amdgcn_s_wait_tensorcnt)
#define HAVE_TDM 1
#endif
#endif
#ifndef HAVE_TDM
#define HAVE_TDM 0
#endif
#if __has_include(<hip/amd_detail/amd_gfx1250_TDM.h>)
#define TDM_6ARG 1          // therock-10.0 headers -> 6-arg builtin
#else
#define TDM_6ARG 0          // ROCm 7.2 -> 5-arg builtin
#endif

static __device__ __forceinline__ __bf16 f2bf(float f) {
    unsigned u = __builtin_bit_cast(unsigned, f);
    unsigned r = u + 0x7FFFu + ((u >> 16) & 1u);     // RNE
    return __builtin_bit_cast(__bf16, (unsigned short)(r >> 16));
}
static __device__ __forceinline__ float bf2f(__bf16 h) {
    unsigned u = ((unsigned)__builtin_bit_cast(unsigned short, h)) << 16;
    return __builtin_bit_cast(float, u);
}
static __device__ __forceinline__ __bf16 bf_zero() {
    return __builtin_bit_cast(__bf16, (unsigned short)0);
}

// ---------------------------------------------------------------------------
// TDM: async-DMA a [tile_d1 x tile_d0] bf16 2D tile into LDS.
//  - tensor_dim0/1 clamp the tile (OOB elements are zero-filled)
//  - LDS padding: pad_amount dwords after every pad_interval dwords
//    (codes: interval = 2<<code dwords, amount = code+1 dwords)
// ---------------------------------------------------------------------------
static __device__ __forceinline__ void tdm_load_2d_bf16(
    const __bf16* gptr, unsigned lds_off,
    unsigned tensor_d0, unsigned tensor_d1,
    unsigned tile_d0, unsigned tile_d1,
    unsigned pad_interval_code, unsigned pad_amount_code,
    unsigned stride_elems)
{
#if HAVE_TDM
    unsigned long long ga = (unsigned long long)(size_t)gptr;
    u32x4 g0;
    g0[0] = 1u;                                       // count=1, user descriptor
    g0[1] = lds_off;                                  // LDS byte address
    g0[2] = (unsigned)ga;                             // global_addr[31:0]
    g0[3] = (unsigned)((ga >> 32) & 0x1FFFFFFull)     // global_addr[56:32]
          | (2u << 30);                               // type = 2 ("image")
    i32x8 g1;
    g1[0] = (int)((1u << 16)                          // data_size = 2B
                | (1u << 20)                          // pad_enable
                | (pad_interval_code << 22)
                | (pad_amount_code << 25));
    g1[1] = (int)((tensor_d0 & 0xFFFFu) << 16);       // tensor_dim0 lo16 @bit48
    g1[2] = (int)((tensor_d0 >> 16) | ((tensor_d1 & 0xFFFFu) << 16));
    g1[3] = (int)((tensor_d1 >> 16) | (tile_d0 << 16));
    g1[4] = (int)tile_d1;                             // tile_dim1 | tile_dim2=0
    g1[5] = (int)stride_elems;                        // tensor_dim0_stride lo32
    g1[6] = 0;                                        // stride hi | dim1_stride
    g1[7] = 0;
    i32x4 z4 = {0, 0, 0, 0};
#if TDM_6ARG
    i32x8 z8 = {0, 0, 0, 0, 0, 0, 0, 0};
    __builtin_amdgcn_tensor_load_to_lds(g0, g1, z4, z4, z8, 0);
#else
    __builtin_amdgcn_tensor_load_to_lds(g0, g1, z4, z4, 0);
#endif
#else
    (void)gptr; (void)lds_off; (void)tensor_d0; (void)tensor_d1;
    (void)tile_d0; (void)tile_d1; (void)pad_interval_code;
    (void)pad_amount_code; (void)stride_elems;
#endif
}

// ---------------------------------------------------------------------------
// bf16 WMMA GEMM:  C[m,n] = act(alpha * sum_k A[m,k]*B + bias[n])
//   AMODE 0: linear A [M,K] bf16 (TDM-staged)
//   AMODE 1: implicit im2col of 3x3 conv over A=[Bimg, HH*WW, Cin] bf16
//   BT 1: B [N,K] bf16 ("NT", VGPR-staged transpose)
//   BT 0: B [K,N] bf16 ("NN", TDM-staged)
//   OUTBF: store bf16 instead of f32
// Tile 128x128x32, 8 wave32 waves, double-buffered LDS, 1 barrier / K-step.
// ---------------------------------------------------------------------------
template<int AMODE, int BT, int RELU, int HASBIAS, int OUTBF>
__global__ __launch_bounds__(256)
void gemm_bf16_wmma(const __bf16* __restrict__ Aptr, const __bf16* __restrict__ Bptr,
                    const float* __restrict__ biasPtr, void* __restrict__ Cout,
                    int M, int N, int K, int lda, int ldb, int ldc,
                    long long sA, long long sB, long long sC,
                    float alpha, int HH, int WW, int Cin)
{
    constexpr int BM = 128, BN = 128, BK = 32;
    constexpr bool A_TDM = (AMODE == 0) && (HAVE_TDM != 0);
    constexpr bool B_TDM = (BT == 0) && (HAVE_TDM != 0);
    __shared__ __bf16 As[2][BM][BK + 8];   // 80B rows  (TDM pad 3/3)
    __shared__ __bf16 Bs[2][BK][BN + 8];   // 272B rows (TDM pad 5/3)

    const int t = threadIdx.x, lane = t & 31, wid = t >> 5;
    const int wm = (wid & 1) * 64, wn = (wid >> 1) * 32;
    const int bz = blockIdx.z;
    const __bf16* A  = Aptr + (size_t)bz * (size_t)sA;
    const __bf16* Bm = Bptr + (size_t)bz * (size_t)sB;
    const int m0 = blockIdx.y * BM, n0 = blockIdx.x * BN;
    const int Simg = HH * WW;

    v8f acc[4][2];
#pragma unroll
    for (int i = 0; i < 4; ++i)
#pragma unroll
        for (int j = 0; j < 2; ++j)
#pragma unroll
            for (int e = 0; e < 8; ++e) acc[i][j][e] = 0.0f;

    auto stageA_tdm = [&](int k0, int buf) {   // one 16-row slice per wave
        const int slice = wid * 16;
        long long rem = (long long)M - (m0 + slice);
        unsigned rows = rem > 0 ? (unsigned)rem : 0u;
        const __bf16* g = A + (size_t)(m0 + slice) * lda + k0;
        unsigned lds_off = (unsigned)(size_t)&As[buf][slice][0];
        // tile 32x16, rows 64B -> pad 16B => 80B pitch
        tdm_load_2d_bf16(g, lds_off, 32u, rows, 32u, 16u, 3u, 3u, (unsigned)lda);
    };
    auto stageB_tdm = [&](int k0, int buf) {   // NN: 4 k-rows per wave
        const int slice = wid * 4;
        long long remN = (long long)N - n0;
        unsigned cols = remN > 0 ? (unsigned)remN : 0u;     // clamps N edge
        unsigned rows = (unsigned)(K - (k0 + slice));       // >= 4 always
        const __bf16* g = Bm + (size_t)(k0 + slice) * ldb + n0;
        unsigned lds_off = (unsigned)(size_t)&Bs[buf][slice][0];
        // tile 128x4, rows 256B -> pad 16B => 272B pitch
        tdm_load_2d_bf16(g, lds_off, cols, rows, 128u, 4u, 5u, 3u, (unsigned)ldb);
    };
    auto loadA_regs = [&](int k0, v4bf* arg) {
#pragma unroll
        for (int i = 0; i < 4; ++i) {
            int e = i * 1024 + t * 4, r = e >> 5, cc = e & 31, gm = m0 + r;
            v4bf v; v[0] = v[1] = v[2] = v[3] = bf_zero();
            if (AMODE == 0) {
                if (gm < M) v = *(const v4bf*)(A + (size_t)gm * lda + (k0 + cc));
            } else {
                int gk = k0 + cc;
                int tap = gk / Cin, ci = gk - tap * Cin;  // 4 elems share a tap
                if (gm < M) {
                    int b  = gm / Simg, p = gm - b * Simg;
                    int ky = tap / 3, kx = tap - ky * 3;
                    int yy = (p / WW) + ky - 1, xx = (p % WW) + kx - 1;
                    if (yy >= 0 && yy < HH && xx >= 0 && xx < WW) {
                        const __bf16* src =
                            A + ((size_t)(b * Simg + yy * WW + xx) * Cin + ci);
                        __builtin_prefetch(src, 0, 1);   // global_prefetch_b8
                        v = *(const v4bf*)src;
                    }
                }
            }
            arg[i] = v;
        }
    };
    auto storeA = [&](int buf, const v4bf* arg) {
#pragma unroll
        for (int i = 0; i < 4; ++i) {
            int e = i * 1024 + t * 4, r = e >> 5, cc = e & 31;
            *(v4bf*)&As[buf][r][cc] = arg[i];
        }
    };
    auto loadB_regs = [&](int k0, v4bf* brg) {   // NT path only
#pragma unroll
        for (int i = 0; i < 4; ++i) {
            v4bf v; v[0] = v[1] = v[2] = v[3] = bf_zero();
            int e = i * 1024 + t * 4;
            int n = e >> 5, kk = e & 31;
            if (n0 + n < N) v = *(const v4bf*)(Bm + (size_t)(n0 + n) * ldb + (k0 + kk));
            brg[i] = v;
        }
    };
    auto storeB = [&](int buf, const v4bf* brg) {  // NT transpose scatter
#pragma unroll
        for (int i = 0; i < 4; ++i) {
            int e = i * 1024 + t * 4;
            int n = e >> 5, kk = e & 31;
#pragma unroll
            for (int j = 0; j < 4; ++j) Bs[buf][kk + j][n] = brg[i][j];
        }
    };

    const int nK = K / BK;
    // ---- prologue: stage tile 0 into buffer 0 ----
    {
        v4bf arg[4], brg[4];
        if (A_TDM) stageA_tdm(0, 0);
        else { loadA_regs(0, arg); storeA(0, arg); }
        if (B_TDM) stageB_tdm(0, 0);
        else { loadB_regs(0, brg); storeB(0, brg); }
#if HAVE_TDM
        if (A_TDM || B_TDM) __builtin_amdgcn_s_wait_tensorcnt(0);
#endif
    }
    __syncthreads();

    for (int kt = 0; kt < nK; ++kt) {
        const int cur = kt & 1, nxt = cur ^ 1;
        const bool more = (kt + 1) < nK;
        v4bf arg[4], brg[4];
        if (more) {                       // overlap next-tile fetch with WMMA
            if (A_TDM) stageA_tdm((kt + 1) * BK, nxt);
            else       loadA_regs((kt + 1) * BK, arg);
            if (B_TDM) stageB_tdm((kt + 1) * BK, nxt);
            else       loadB_regs((kt + 1) * BK, brg);
        }

        // ---- fragments from current buffers + 8 WMMAs ----
        const int half = lane >> 4, mr = lane & 15;
        v16bf af[4], bfr[2];
#pragma unroll
        for (int mt = 0; mt < 4; ++mt) {
            const v8bf lo = *(const v8bf*)&As[cur][wm + mt * 16 + mr][half * 8];
            const v8bf hi = *(const v8bf*)&As[cur][wm + mt * 16 + mr][half * 8 + 16];
            af[mt] = __builtin_shufflevector(lo, hi, 0,1,2,3,4,5,6,7,8,9,10,11,12,13,14,15);
        }
#pragma unroll
        for (int nt = 0; nt < 2; ++nt) {
#pragma unroll
            for (int s = 0; s < 16; ++s)
                bfr[nt][s] = Bs[cur][half * 8 + (s & 7) + ((s >> 3) << 4)][wn + nt * 16 + mr];
        }
#pragma unroll
        for (int mt = 0; mt < 4; ++mt)
#pragma unroll
            for (int nt = 0; nt < 2; ++nt)
                acc[mt][nt] = __builtin_amdgcn_wmma_f32_16x16x32_bf16(
                    false, af[mt], false, bfr[nt], (short)0, acc[mt][nt], false, false);

        if (more) {
            if (!A_TDM) storeA(nxt, arg);
            if (!B_TDM) storeB(nxt, brg);
#if HAVE_TDM
            if (A_TDM || B_TDM) __builtin_amdgcn_s_wait_tensorcnt(0);
#endif
        }
        __syncthreads();
    }

    // ---- epilogue ----
    const int half = lane >> 4, cn = lane & 15;
    float*  Cf = (float*)Cout  + (size_t)bz * (size_t)sC;
    __bf16* Ch = (__bf16*)Cout + (size_t)bz * (size_t)sC;
#pragma unroll
    for (int mt = 0; mt < 4; ++mt)
#pragma unroll
        for (int nt = 0; nt < 2; ++nt)
#pragma unroll
            for (int r = 0; r < 8; ++r) {
                int gm = m0 + wm + mt * 16 + (half << 3) + r;
                int gn = n0 + wn + nt * 16 + cn;
                if (gm < M && gn < N) {
                    float v = alpha * acc[mt][nt][r];
                    if (HASBIAS) v += biasPtr[gn];
                    if (RELU)    v = v > 0.f ? v : 0.f;
                    if (OUTBF) Ch[(size_t)gm * ldc + gn] = f2bf(v);
                    else       Cf[(size_t)gm * ldc + gn] = v;
                }
            }
}

// ---------------------------------------------------------------------------
// Batched 32x32 transpose, output bf16. INBF selects f32/bf16 input.
// ---------------------------------------------------------------------------
template<int INBF>
__global__ __launch_bounds__(256)
void transpose_to_bf16(const void* __restrict__ in, __bf16* __restrict__ outp, int R, int CC)
{
    __shared__ float tile[32][33];
    const size_t bs = (size_t)R * CC;
    int c0 = blockIdx.x * 32, r0 = blockIdx.y * 32;
    int tx = threadIdx.x & 31, ty = threadIdx.x >> 5;
#pragma unroll
    for (int j = 0; j < 32; j += 8) {
        int r = r0 + ty + j, c = c0 + tx;
        float v = 0.f;
        if (r < R && c < CC) {
            if (INBF) v = bf2f(((const __bf16*)in)[(size_t)blockIdx.z * bs + (size_t)r * CC + c]);
            else      v = ((const float*)in)[(size_t)blockIdx.z * bs + (size_t)r * CC + c];
        }
        tile[ty + j][tx] = v;
    }
    __syncthreads();
#pragma unroll
    for (int j = 0; j < 32; j += 8) {
        int c = c0 + ty + j, r = r0 + tx;
        if (c < CC && r < R)
            outp[(size_t)blockIdx.z * bs + (size_t)c * R + r] = f2bf(tile[tx][ty + j]);
    }
}

__global__ void cvt_f32_bf16_kernel(const float* __restrict__ in, __bf16* __restrict__ out,
                                    long long n)
{
    long long gid = (long long)blockIdx.x * 256 + threadIdx.x;
    if (gid < n) out[gid] = f2bf(in[gid]);
}

// w2[co][ci][ky][kx] -> w2r[co][tap][ci] (bf16)
__global__ void repack_w2_kernel(const float* __restrict__ w2, __bf16* __restrict__ w2r,
                                 int Cout, int Cin)
{
    int gid = blockIdx.x * blockDim.x + threadIdx.x;
    if (gid >= Cout * Cin * 9) return;
    int co = gid / (Cin * 9), rem = gid - co * (Cin * 9);
    int tap = rem / Cin, ci = rem - tap * Cin;
    w2r[(size_t)co * Cin * 9 + (size_t)tap * Cin + ci] =
        f2bf(w2[(size_t)co * Cin * 9 + (size_t)ci * 9 + tap]);
}

// softmax over query axis + hard one-hot straight-through; bf16 output
__global__ __launch_bounds__(256)
void assign_softmax_kernel(const float* __restrict__ raw, __bf16* __restrict__ attn,
                           int Bv, int NQv, int Sv)
{
    int gid = blockIdx.x * blockDim.x + threadIdx.x;
    if (gid >= Bv * Sv) return;
    int b = gid / Sv, s = gid - b * Sv;
    const float* rp = raw + (size_t)b * NQv * Sv + s;
    float mx = -3.4e38f; int am = 0;
    for (int n = 0; n < NQv; ++n) {
        float v = rp[(size_t)n * Sv];
        if (v > mx) { mx = v; am = n; }
    }
    float den = 0.f;
    for (int n = 0; n < NQv; ++n) den += __expf(rp[(size_t)n * Sv] - mx);
    float inv = 1.f / den;
    __bf16* ap = attn + (size_t)b * NQv * Sv + s;
    for (int n = 0; n < NQv; ++n) {
        float soft = __expf(rp[(size_t)n * Sv] - mx) * inv;
        float hard = (n == am) ? 1.f : 0.f;
        ap[(size_t)n * Sv] = f2bf((hard - soft) + soft);
    }
}

__global__ __launch_bounds__(256)
void rowsum_inv_kernel(const __bf16* __restrict__ attn, float* __restrict__ invden, int Sv)
{
    __shared__ float red[256];
    const __bf16* ap = attn + (size_t)blockIdx.x * Sv;
    float s = 0.f;
    for (int i = threadIdx.x; i < Sv; i += 256) s += bf2f(ap[i]);
    red[threadIdx.x] = s;
    __syncthreads();
    for (int st = 128; st > 0; st >>= 1) {
        if ((int)threadIdx.x < st) red[threadIdx.x] += red[threadIdx.x + st];
        __syncthreads();
    }
    if (threadIdx.x == 0) invden[blockIdx.x] = 1.f / (red[0] + 1.f);
}

// out_bf[row,c] = f32_in[row,c] * invden[row]
__global__ void scale_cvt_kernel(const float* __restrict__ x, const float* __restrict__ invden,
                                 __bf16* __restrict__ out, int Ccols, int total)
{
    int gid = blockIdx.x * blockDim.x + threadIdx.x;
    if (gid < total) out[gid] = f2bf(x[gid] * invden[gid / Ccols]);
}

// d_out[nq, b, c] = fin[b*NQ+nq, c] + queryT[b*NQ+nq, c]
__global__ void assemble_out_kernel(const float* __restrict__ fin, const __bf16* __restrict__ qT,
                                    float* __restrict__ out, int NQv, int Bv, int Cv)
{
    int gid = blockIdx.x * blockDim.x + threadIdx.x;
    if (gid >= NQv * Bv * Cv) return;
    int c = gid % Cv, rem = gid / Cv, b = rem % Bv, nq = rem / Bv;
    size_t src = ((size_t)b * NQv + nq) * Cv + c;
    out[gid] = fin[src] + bf2f(qT[src]);
}

// ---------------------------------------------------------------------------
extern "C" void kernel_launch(void* const* d_in, const int* in_sizes, int n_in,
                              void* d_out, int out_size, void* d_ws, size_t ws_size,
                              hipStream_t stream)
{
    const float* x   = (const float*)d_in[0];
    const float* w1  = (const float*)d_in[1];
    const float* b1  = (const float*)d_in[2];
    const float* w2  = (const float*)d_in[3];
    const float* b2  = (const float*)d_in[4];
    const float* w3  = (const float*)d_in[5];
    const float* b3  = (const float*)d_in[6];
    const float* mw1 = (const float*)d_in[7];
    const float* mb1 = (const float*)d_in[8];
    const float* mw2 = (const float*)d_in[9];
    const float* mb2 = (const float*)d_in[10];
    const float* mw3 = (const float*)d_in[11];
    const float* mb3 = (const float*)d_in[12];
    const float* qw  = (const float*)d_in[13];
    const float* qb  = (const float*)d_in[14];
    const float* kw  = (const float*)d_in[15];
    const float* kb  = (const float*)d_in[16];
    const float* vw  = (const float*)d_in[17];
    const float* vb  = (const float*)d_in[18];
    const float* pw  = (const float*)d_in[19];
    const float* pb  = (const float*)d_in[20];
    (void)in_sizes; (void)n_in; (void)out_size; (void)ws_size;
    float* out = (float*)d_out;

    const int Bn = 8, CIN = 256, Hh = 64, Ww = 64, Cch = 256, S = 4096, NQ = 100;
    const int MS = Bn * S;
    const size_t SZ_BIG = (size_t)MS * 256;          // 8,388,608 elements

    // bump allocator over d_ws (256B aligned)
    char* base = (char*)d_ws;
    size_t off = 0;
    auto alloc = [&](size_t bytes) -> void* {
        void* p = base + off;
        off += (bytes + 255) & ~(size_t)255;
        return p;
    };
    __bf16* R0   = (__bf16*)alloc(SZ_BIG * 2);        // xT -> h2 -> k
    __bf16* R1   = (__bf16*)alloc(SZ_BIG * 2);        // h1 -> feat -> vT
    __bf16* R2   = (__bf16*)alloc(SZ_BIG * 2);        // featT -> v
    __bf16* w1b  = (__bf16*)alloc(65536 * 2);
    __bf16* w2rb = (__bf16*)alloc(589824 * 2);
    __bf16* w3b  = (__bf16*)alloc(65536 * 2);
    __bf16* mw1b = (__bf16*)alloc(1048576 * 2);
    __bf16* mw2b = (__bf16*)alloc(65536 * 2);
    __bf16* mw3b = (__bf16*)alloc(25600 * 2);
    __bf16* qwb  = (__bf16*)alloc(65536 * 2);
    __bf16* kwb  = (__bf16*)alloc(65536 * 2);
    __bf16* vwb  = (__bf16*)alloc(65536 * 2);
    __bf16* pwb  = (__bf16*)alloc(65536 * 2);
    __bf16* m1b  = (__bf16*)alloc(524288 * 2);
    __bf16* m2b  = (__bf16*)alloc(524288 * 2);
    __bf16* m3b  = (__bf16*)alloc(204800 * 2);
    __bf16* qTb  = (__bf16*)alloc(204800 * 2);
    __bf16* qprj = (__bf16*)alloc(204800 * 2);
    float*  raw  = (float*)alloc((size_t)Bn * NQ * S * 4);
    __bf16* attn = (__bf16*)alloc((size_t)Bn * NQ * S * 2);
    float*  invd = (float*)alloc(800 * 4);
    float*  av   = (float*)alloc(204800 * 4);
    __bf16* avb  = (__bf16*)alloc(204800 * 2);
    float*  fin  = (float*)alloc(204800 * 4);

    // ---- one-time bf16 conversion of weights ----
    cvt_f32_bf16_kernel<<<256,  256, 0, stream>>>(w1,  w1b,  65536);
    repack_w2_kernel<<<(589824 + 255) / 256, 256, 0, stream>>>(w2, w2rb, 256, 256);
    cvt_f32_bf16_kernel<<<256,  256, 0, stream>>>(w3,  w3b,  65536);
    cvt_f32_bf16_kernel<<<4096, 256, 0, stream>>>(mw1, mw1b, 1048576);
    cvt_f32_bf16_kernel<<<256,  256, 0, stream>>>(mw2, mw2b, 65536);
    cvt_f32_bf16_kernel<<<100,  256, 0, stream>>>(mw3, mw3b, 25600);
    cvt_f32_bf16_kernel<<<256,  256, 0, stream>>>(qw,  qwb,  65536);
    cvt_f32_bf16_kernel<<<256,  256, 0, stream>>>(kw,  kwb,  65536);
    cvt_f32_bf16_kernel<<<256,  256, 0, stream>>>(vw,  vwb,  65536);
    cvt_f32_bf16_kernel<<<256,  256, 0, stream>>>(pw,  pwb,  65536);

    // 1. xT[b,p,ci] (bf16)
    transpose_to_bf16<0><<<dim3(128, 8, Bn), 256, 0, stream>>>(x, R0, CIN, S);
    // 2. conv1 1x1 + relu -> h1 (bf16)
    gemm_bf16_wmma<0,1,1,1,1><<<dim3(2, 256, 1), 256, 0, stream>>>(
        R0, w1b, b1, R1, MS, 256, CIN, CIN, CIN, 256, 0, 0, 0, 1.f, 0, 0, 0);
    // 3. conv2 3x3 implicit im2col + relu -> h2 (bf16)
    gemm_bf16_wmma<1,1,1,1,1><<<dim3(2, 256, 1), 256, 0, stream>>>(
        R1, w2rb, b2, R0, MS, 256, 2304, 0, 2304, 256, 0, 0, 0, 1.f, Hh, Ww, 256);
    // 4. conv3 1x1 -> feat [B,S,C] (bf16)
    gemm_bf16_wmma<0,1,0,1,1><<<dim3(2, 256, 1), 256, 0, stream>>>(
        R0, w3b, b3, R1, MS, Cch, 256, 256, 256, Cch, 0, 0, 0, 1.f, 0, 0, 0);
    // 5. featT [B,C,S]
    transpose_to_bf16<1><<<dim3(8, 128, Bn), 256, 0, stream>>>(R1, R2, S, Cch);
    // 6-8. query mixer MLP (NN weights, TDM-staged B)
    gemm_bf16_wmma<0,0,1,1,1><<<dim3(2, 16, 1), 256, 0, stream>>>(
        R2, mw1b, mb1, m1b, Bn * Cch, 256, S, S, 256, 256, 0, 0, 0, 1.f, 0, 0, 0);
    gemm_bf16_wmma<0,0,1,1,1><<<dim3(2, 16, 1), 256, 0, stream>>>(
        m1b, mw2b, mb2, m2b, Bn * Cch, 256, 256, 256, 256, 256, 0, 0, 0, 1.f, 0, 0, 0);
    gemm_bf16_wmma<0,0,0,1,1><<<dim3(1, 16, 1), 256, 0, stream>>>(
        m2b, mw3b, mb3, m3b, Bn * Cch, NQ, 256, 256, NQ, NQ, 0, 0, 0, 1.f, 0, 0, 0);
    // 9. queryT [B,NQ,C]
    transpose_to_bf16<1><<<dim3(4, 8, Bn), 256, 0, stream>>>(m3b, qTb, Cch, NQ);
    // 10. q projection
    gemm_bf16_wmma<0,0,0,1,1><<<dim3(2, 7, 1), 256, 0, stream>>>(
        qTb, qwb, qb, qprj, Bn * NQ, Cch, Cch, Cch, Cch, Cch, 0, 0, 0, 1.f, 0, 0, 0);
    // 11-12. k / v projections
    gemm_bf16_wmma<0,0,0,1,1><<<dim3(2, 256, 1), 256, 0, stream>>>(
        R1, kwb, kb, R0, MS, Cch, Cch, Cch, Cch, Cch, 0, 0, 0, 1.f, 0, 0, 0);
    gemm_bf16_wmma<0,0,0,1,1><<<dim3(2, 256, 1), 256, 0, stream>>>(
        R1, vwb, vb, R2, MS, Cch, Cch, Cch, Cch, Cch, 0, 0, 0, 1.f, 0, 0, 0);
    // 13. vT [B,C,S]
    transpose_to_bf16<1><<<dim3(8, 128, Bn), 256, 0, stream>>>(R2, R1, S, Cch);
    // 14. raw = (q @ k^T) / 16, batched (f32 out)
    gemm_bf16_wmma<0,1,0,0,0><<<dim3(32, 1, Bn), 256, 0, stream>>>(
        qprj, R0, nullptr, raw, NQ, S, Cch, Cch, Cch, S,
        (long long)NQ * Cch, (long long)S * Cch, (long long)NQ * S, 0.0625f, 0, 0, 0);
    // 15. softmax-over-queries + hard assignment (bf16 out)
    assign_softmax_kernel<<<(Bn * S + 255) / 256, 256, 0, stream>>>(raw, attn, Bn, NQ, S);
    // 16. 1/(rowsum+1)
    rowsum_inv_kernel<<<Bn * NQ, 256, 0, stream>>>(attn, invd, S);
    // 17. av = attn @ v (vT as NT operand), batched (f32 out)
    gemm_bf16_wmma<0,1,0,0,0><<<dim3(2, 1, Bn), 256, 0, stream>>>(
        attn, R1, nullptr, av, NQ, Cch, S, S, S, Cch,
        (long long)NQ * S, (long long)Cch * S, (long long)NQ * Cch, 1.f, 0, 0, 0);
    // 18. normalize rows, emit bf16
    scale_cvt_kernel<<<800, 256, 0, stream>>>(av, invd, avb, Cch, Bn * NQ * Cch);
    // 19. output projection (f32 out)
    gemm_bf16_wmma<0,0,0,1,0><<<dim3(2, 7, 1), 256, 0, stream>>>(
        avb, pwb, pb, fin, Bn * NQ, Cch, Cch, Cch, Cch, Cch, 0, 0, 0, 1.f, 0, 0, 0);
    // 20. residual + [NQ,B,C]
    assemble_out_kernel<<<(NQ * Bn * Cch + 255) / 256, 256, 0, stream>>>(
        fin, qTb, out, NQ, Bn, Cch);
}
